// MeshCaster_70901320122393
// MI455X (gfx1250) — matrix operand order repaired and checked by reading, not compiled, not run on hardware
//
#include <hip/hip_runtime.h>

// ---------------- problem constants ----------------
#define NSAMP   32768
#define NMESH   8
#define NROWS   (NSAMP * NMESH)     // 262144
#define NC      256
#define NVERT1  50001               // N_VERTS + 1
#define ROWS_PER_BLOCK 32
#define THREADS 64                  // 2 wave32 per block

typedef __bf16 bf16;
typedef bf16     v16bf  __attribute__((ext_vector_type(16)));
typedef float    v8f    __attribute__((ext_vector_type(8)));
typedef unsigned u32x4  __attribute__((ext_vector_type(4)));
typedef float    f32x4  __attribute__((ext_vector_type(4)));
typedef int      i32x4v __attribute__((ext_vector_type(4)));
typedef int      i32x8v __attribute__((ext_vector_type(8)));

#define LDSA_STRIDE 264             // bf16 elems per activation row (528B, 16B aligned, conflict-free)
#define LDSW_STRIDE 40              // bf16 elems per weight-slab row (80B, 16B aligned)

// ---------------- workspace layout (bytes) ----------------
#define WS_WT_PROJ   0                         // bf16 [256][64]
#define WS_WT_V0     (WS_WT_PROJ + 256*64*2)   // bf16 [256][256]
#define WS_WT_V1     (WS_WT_V0 + 256*256*2)
#define WS_WT_T0     (WS_WT_V1 + 256*256*2)
#define WS_WT_T1     (WS_WT_T0 + 256*256*2)
#define WS_ALPHA_WC  (WS_WT_T1 + 256*256*2)    // f32 [256]
#define WS_ALPHA_B   (WS_ALPHA_WC + 256*4)     // f32 [1] (+pad)
#define WS_COLOR_WC  (WS_ALPHA_B + 16)         // f32 [512*3]
#define WS_COLOR_B   (WS_COLOR_WC + 512*3*4)   // f32 [3]

// ---------------- prep: transpose weights to bf16 [N][Kpad] ----------------
__global__ void k_transpose_bf16(const float* __restrict__ src, bf16* __restrict__ dst,
                                 int K, int N, int Kpad) {
  int i = blockIdx.x * blockDim.x + threadIdx.x;
  if (i >= N * Kpad) return;
  int n = i / Kpad, k = i % Kpad;
  float v = (k < K) ? src[k * N + n] : 0.0f;
  dst[i] = (bf16)v;
}

// ---------------- prep: collapse identity-activation heads ----------------
__global__ void k_collapse_heads(const float* __restrict__ aW1, const float* __restrict__ ab1,
                                 const float* __restrict__ aW2, const float* __restrict__ ab2,
                                 const float* __restrict__ cW1, const float* __restrict__ cb1,
                                 const float* __restrict__ cW2, const float* __restrict__ cb2,
                                 float* __restrict__ alphaWc, float* __restrict__ alphaB,
                                 float* __restrict__ colorWc, float* __restrict__ colorB) {
  int i = blockIdx.x * blockDim.x + threadIdx.x;
  if (i < 256) {                       // alphaWc[k] = sum_j aW1[k][j] * aW2[j]
    float s = 0.f;
    for (int j = 0; j < 256; ++j) s += aW1[i * 256 + j] * aW2[j];
    alphaWc[i] = s;
  } else if (i < 256 + 512 * 3) {      // colorWc[k][c] = sum_j cW1[k][j] * cW2[j][c]
    int t = i - 256, k = t / 3, c = t % 3;
    float s = 0.f;
    for (int j = 0; j < 512; ++j) s += cW1[k * 512 + j] * cW2[j * 3 + c];
    colorWc[k * 3 + c] = s;
  } else if (i == 256 + 512 * 3) {     // alphaB = ab1 . aW2 + ab2
    float s = ab2[0];
    for (int j = 0; j < 256; ++j) s += ab1[j] * aW2[j];
    alphaB[0] = s;
  } else if (i < 256 + 512 * 3 + 4) {  // colorB[c] = cb1 . cW2[:,c] + cb2[c]
    int c = i - (256 + 512 * 3 + 1);
    float s = cb2[c];
    for (int j = 0; j < 512; ++j) s += cb1[j] * cW2[j * 3 + c];
    colorB[c] = s;
  }
}

// ---------------- wave helpers ----------------
__device__ __forceinline__ float wred32(float v) {
  v += __shfl_xor(v, 1, 32);  v += __shfl_xor(v, 2, 32);
  v += __shfl_xor(v, 4, 32);  v += __shfl_xor(v, 8, 32);
  v += __shfl_xor(v, 16, 32);
  return v;
}
__device__ __forceinline__ float wred16(float v) {
  v += __shfl_xor(v, 1, 32);  v += __shfl_xor(v, 2, 32);
  v += __shfl_xor(v, 4, 32);  v += __shfl_xor(v, 8, 32);
  return v;
}

// A fragment: 16x32 bf16, lane l holds row (l&15); lanes<16: K=[kk,kk+8)+[kk+16,kk+24),
// lanes>=16: K=[kk+8,kk+16)+[kk+24,kk+32)  (ISA 7.12.2 16-bit A layout)
__device__ __forceinline__ v16bf load_a_frag(const bf16* actrow, int kk, int half) {
  union { v16bf v; u32x4 q[2]; } u;
  const u32x4* p = (const u32x4*)(actrow + kk + half * 8);
  u.q[0] = p[0];
  u.q[1] = p[2];          // +16 bf16 elems
  return u.v;
}
// B fragment: 32x16 bf16, lane l covers N=(l&15); K=[khalf,khalf+8)+[khalf+16,khalf+24), khalf=8*(l>>4).
// Weights stored transposed [N][K] so this is 2 contiguous b128 LDS loads.
__device__ __forceinline__ v16bf load_b_frag(const bf16* ldsw, int n, int khalf) {
  union { v16bf v; u32x4 q[2]; } u;
  const u32x4* p = (const u32x4*)(ldsw + n * LDSW_STRIDE + khalf);
  u.q[0] = p[0];
  u.q[1] = p[2];
  return u.v;
}

__device__ __forceinline__ void zero_acc(v8f (&acc)[16]) {
#pragma unroll
  for (int nt = 0; nt < 16; ++nt)
    acc[nt] = (v8f){0.f, 0.f, 0.f, 0.f, 0.f, 0.f, 0.f, 0.f};
}

// ---------------- Tensor Data Mover: async slab copy (global -> LDS) ----------------
// Copies a 2D tile [256 rows][32 bf16] of a [256][ktot] bf16 matrix into LDS,
// inserting 4 DWORDs of padding every 16 DWORDs (one tile row), giving the
// LDSW_STRIDE=40 element row pitch. Tracked by TENSORcnt.
__device__ __forceinline__ void tdm_load_slab(const bf16* gsrc, int ktot, bf16* ldsdst) {
  unsigned long long ga = (unsigned long long)(size_t)gsrc;
  unsigned lds_addr = (unsigned)(size_t)ldsdst;      // flat LDS aperture: low 32 bits = LDS byte addr
  u32x4 g0;
  g0.x = 1u;                                          // count=1, is_restore=0, gather off
  g0.y = lds_addr;                                    // [63:32]  lds_addr
  g0.z = (unsigned)(ga & 0xffffffffull);              // [95:64]  global_addr lo
  g0.w = (unsigned)((ga >> 32) & 0x01ffffffull)       // [120:96] global_addr hi
         | (2u << 30);                                // [127:126] type = 2 (image/tensor)
  i32x8v g1;
  g1[0] = (1 << 16)                                   // data_size = 2 bytes
        | (1 << 20)                                   // pad_enable
        | (3 << 22)                                   // pad_interval: 16 DWORDs
        | (3 << 25);                                  // pad_amount:   4 DWORDs
  g1[1] = (ktot & 0xffff) << 16;                      // tensor_dim0[15:0]
  g1[2] = (256 << 16);                                // tensor_dim0 hi=0 | tensor_dim1[15:0]=256
  g1[3] = (32 << 16);                                 // tensor_dim1 hi=0 | tile_dim0=32
  g1[4] = 256;                                        // tile_dim1=256, tile_dim2=0
  g1[5] = ktot;                                       // tensor_dim0_stride lo
  g1[6] = 0;                                          // stride0 hi | stride1 lo
  g1[7] = 0;                                          // stride1 hi
  i32x4v z4 = (i32x4v){0, 0, 0, 0};
#if defined(__clang_major__) && (__clang_major__ >= 23)
  i32x8v z8 = (i32x8v){0, 0, 0, 0, 0, 0, 0, 0};
  __builtin_amdgcn_tensor_load_to_lds(g0, g1, z4, z4, z8, 0);
#else
  __builtin_amdgcn_tensor_load_to_lds(g0, g1, z4, z4, 0);
#endif
}

// One 256-wide output layer: TDM-staged, double-buffered K-slabs; each wave
// computes a 16-row x 256-col tile (16 accumulators of 16x16) with bf16 WMMA.
__device__ __forceinline__ void gemm_layer_tdm(const bf16* __restrict__ wt, int ktot,
                                               const bf16* lds_act,
                                               bf16* lds_w0, bf16* lds_w1,
                                               v8f (&acc)[16]) {
  const int tid   = threadIdx.x;
  const int lane  = tid & 31;
  const int wv    = tid >> 5;
  const int half  = lane >> 4;
  const int khalf = half * 16;
  const int nlo   = lane & 15;
  const bf16* actrow = lds_act + (wv * 16 + nlo) * LDSA_STRIDE;
  const int nsteps = ktot >> 5;

  if (wv == 0) tdm_load_slab(wt, ktot, lds_w0);       // prologue: slab 0
  for (int i = 0; i < nsteps; ++i) {
    const bf16* cur = (i & 1) ? lds_w1 : lds_w0;
    if (wv == 0) {
      if (i + 1 < nsteps) {
        tdm_load_slab(wt + (i + 1) * 32, ktot, (i & 1) ? lds_w0 : lds_w1);
        __builtin_amdgcn_s_wait_tensorcnt(1);         // slab i landed, slab i+1 in flight
      } else {
        __builtin_amdgcn_s_wait_tensorcnt(0);
      }
    }
    __syncthreads();                                  // publish slab i to both waves
    v16bf a = load_a_frag(actrow, i * 32, half);
#pragma unroll
    for (int nt = 0; nt < 16; ++nt) {
      v16bf b = load_b_frag(cur, nt * 16 + nlo, khalf / 2);
      acc[nt] = __builtin_amdgcn_wmma_f32_16x16x32_bf16(
          false, a, false, b, (short)0, acc[nt], false, false);
    }
    __syncthreads();                                  // WAR: done with slab i before overwrite
  }
}

// bias + ReLU + store back to LDS activation buffer
// (C/D layout: lane l VGPR j -> M=j+8*(l>>4), N=16*nt+(l&15))
__device__ __forceinline__ void store_act_relu(v8f (&acc)[16], bf16* lds_act,
                                               const float* __restrict__ bias) {
  const int lane = threadIdx.x & 31;
  const int wv   = threadIdx.x >> 5;
  const int h    = lane >> 4;
  const int nlo  = lane & 15;
#pragma unroll
  for (int nt = 0; nt < 16; ++nt) {
    int n = nt * 16 + nlo;
    float bv = bias[n];
#pragma unroll
    for (int j = 0; j < 8; ++j) {
      int row = wv * 16 + 8 * h + j;
      lds_act[row * LDSA_STRIDE + n] = (bf16)fmaxf(acc[nt][j] + bv, 0.f);
    }
  }
}

// ---------------- fused main kernel ----------------
__global__ __launch_bounds__(THREADS) void mesh_fused(
    const int* __restrict__ verts, const float* __restrict__ barys,
    const float* __restrict__ views, const float* __restrict__ emb,
    const bf16* __restrict__ wt_proj, const bf16* __restrict__ wt_v0,
    const bf16* __restrict__ wt_v1, const bf16* __restrict__ wt_t0,
    const bf16* __restrict__ wt_t1,
    const float* __restrict__ b_proj, const float* __restrict__ view_b,
    const float* __restrict__ vert_b,
    const float* __restrict__ alphaWc, const float* __restrict__ alphaB,
    const float* __restrict__ colorWc, const float* __restrict__ colorB,
    float* __restrict__ out) {
  __shared__ bf16  lds_X[ROWS_PER_BLOCK * LDSA_STRIDE];   // shared activation buffer
  __shared__ bf16  lds_W[2 * 256 * LDSW_STRIDE];          // double-buffered TDM weight slabs
  __shared__ float lds_cx[ROWS_PER_BLOCK * 3];            // color contribution from embeds

  bf16* lds_W0 = lds_W;
  bf16* lds_W1 = lds_W + 256 * LDSW_STRIDE;

  const int tid  = threadIdx.x;
  const int lane = tid & 31;
  const int wv   = tid >> 5;
  const int rowbase = blockIdx.x * ROWS_PER_BLOCK;

  // ---- phase 1: gather + max_norm renorm + bary blend (each wave: its 16 rows) ----
  {
    const int ch = lane * 8;                 // 32 lanes x 8 floats = 256 channels
    for (int rr = 0; rr < 16; ++rr) {
      const int rl = wv * 16 + rr;
      const int r  = rowbase + rl;
      const int mesh = r & (NMESH - 1);
      float a8[8] = {0, 0, 0, 0, 0, 0, 0, 0};
#pragma unroll
      for (int vtx = 0; vtx < 3; ++vtx) {
        int vid = verts[r * 3 + vtx] + 1;
        const float* erow = emb + ((size_t)mesh * NVERT1 + (size_t)vid) * NC + ch;
        f32x4 x0 = ((const f32x4*)erow)[0];
        f32x4 x1 = ((const f32x4*)erow)[1];
        float ss = x0.x * x0.x + x0.y * x0.y + x0.z * x0.z + x0.w * x0.w +
                   x1.x * x1.x + x1.y * x1.y + x1.z * x1.z + x1.w * x1.w;
        ss = wred32(ss);
        float nrm = sqrtf(ss);
        float scale = (nrm > 1.f) ? (1.f / fmaxf(nrm, 1e-7f)) : 1.f;
        float w = barys[r * 3 + vtx] * scale;
        a8[0] += x0.x * w; a8[1] += x0.y * w; a8[2] += x0.z * w; a8[3] += x0.w * w;
        a8[4] += x1.x * w; a8[5] += x1.y * w; a8[6] += x1.z * w; a8[7] += x1.w * w;
      }
      // color head contribution from embed half of the concat (rows 256..511 of cW1@cW2)
#pragma unroll
      for (int c = 0; c < 3; ++c) {
        float s = 0.f;
#pragma unroll
        for (int i = 0; i < 8; ++i) s += a8[i] * colorWc[(256 + ch + i) * 3 + c];
        s = wred32(s);
        if (lane == 0) lds_cx[rl * 3 + c] = s;
      }
      union { u32x4 q; bf16 hh[8]; } pk;
#pragma unroll
      for (int i = 0; i < 8; ++i) pk.hh[i] = (bf16)a8[i];
      *(u32x4*)(lds_X + rl * LDSA_STRIDE + ch) = pk.q;
    }
  }

  v8f acc[16];

  // ---- vert branch first (so lds_X can be reused by the view branch) ----
  zero_acc(acc);
  gemm_layer_tdm(wt_t0, 256, lds_X, lds_W0, lds_W1, acc);
  store_act_relu(acc, lds_X, vert_b);                 // layer-0 bias
  zero_acc(acc);
  gemm_layer_tdm(wt_t1, 256, lds_X, lds_W0, lds_W1, acc);

  // collapsed alpha head directly from accumulators
  float areg[8] = {0, 0, 0, 0, 0, 0, 0, 0};
  {
    const int nlo = lane & 15;
#pragma unroll
    for (int nt = 0; nt < 16; ++nt) {
      int n = nt * 16 + nlo;
      float bv = vert_b[256 + n];                     // layer-1 bias
      float w = alphaWc[n];
#pragma unroll
      for (int j = 0; j < 8; ++j) areg[j] += fmaxf(acc[nt][j] + bv, 0.f) * w;
    }
#pragma unroll
    for (int j = 0; j < 8; ++j) areg[j] = wred16(areg[j]);
  }

  // ---- sincos view features into lds_X (embeds are dead now), zero-pad K 36->64 ----
  __syncthreads();                                    // all reads of embeds complete
  if (tid < ROWS_PER_BLOCK) {
    const int r = rowbase + tid;
    float xv[3] = {views[r * 3 + 0], views[r * 3 + 1], views[r * 3 + 2]};
#pragma unroll
    for (int i = 0; i < 64; ++i) {
      float val = 0.f;
      if (i < 36) {
        int level = i / 6, rem = i % 6;
        float fx = (float)(1 << level) * xv[rem % 3];
        val = (rem < 3) ? __sinf(fx) : __cosf(fx);
      }
      lds_X[tid * LDSA_STRIDE + i] = (bf16)val;
    }
  }

  // ---- view branch: (pad36->64)x256 proj, then 2x 256x256 + ReLU ----
  zero_acc(acc);
  gemm_layer_tdm(wt_proj, 64, lds_X, lds_W0, lds_W1, acc);
  store_act_relu(acc, lds_X, b_proj);
  zero_acc(acc);
  gemm_layer_tdm(wt_v0, 256, lds_X, lds_W0, lds_W1, acc);
  store_act_relu(acc, lds_X, view_b);                 // layer-0 bias
  zero_acc(acc);
  gemm_layer_tdm(wt_v1, 256, lds_X, lds_W0, lds_W1, acc);

  // color head, view half (rows 0..255 of cW1@cW2), directly from accumulators
  float creg[8][3];
#pragma unroll
  for (int j = 0; j < 8; ++j) { creg[j][0] = 0.f; creg[j][1] = 0.f; creg[j][2] = 0.f; }
  {
    const int nlo = lane & 15;
#pragma unroll
    for (int nt = 0; nt < 16; ++nt) {
      int n = nt * 16 + nlo;
      float bv = view_b[256 + n];                     // layer-1 bias
      float w0 = colorWc[n * 3 + 0], w1 = colorWc[n * 3 + 1], w2 = colorWc[n * 3 + 2];
#pragma unroll
      for (int j = 0; j < 8; ++j) {
        float y = fmaxf(acc[nt][j] + bv, 0.f);        // final ReLU
        creg[j][0] += y * w0; creg[j][1] += y * w1; creg[j][2] += y * w2;
      }
    }
#pragma unroll
    for (int j = 0; j < 8; ++j) {
      creg[j][0] = wred16(creg[j][0]);
      creg[j][1] = wred16(creg[j][1]);
      creg[j][2] = wred16(creg[j][2]);
    }
  }

  // ---- output: lanes 0 and 16 of each wave own rows {8h+j} ----
  if ((lane & 15) == 0) {
    const int h = lane >> 4;
    float cb0 = colorB[0], cb1 = colorB[1], cb2 = colorB[2];
    float ab = alphaB[0];
#pragma unroll
    for (int j = 0; j < 8; ++j) {
      int rl = wv * 16 + 8 * h + j;
      int r  = rowbase + rl;
      f32x4 o;
      o.x = creg[j][0] + lds_cx[rl * 3 + 0] + cb0;
      o.y = creg[j][1] + lds_cx[rl * 3 + 1] + cb1;
      o.z = creg[j][2] + lds_cx[rl * 3 + 2] + cb2;
      o.w = areg[j] + ab;
      *(f32x4*)(out + (size_t)r * 4) = o;
    }
  }
}

// ---------------- host launcher ----------------
extern "C" void kernel_launch(void* const* d_in, const int* in_sizes, int n_in,
                              void* d_out, int out_size, void* d_ws, size_t ws_size,
                              hipStream_t stream) {
  const int*   verts  = (const int*)d_in[0];
  const float* barys  = (const float*)d_in[1];
  const float* views  = (const float*)d_in[2];
  const float* emb    = (const float*)d_in[3];
  const float* w_proj = (const float*)d_in[4];
  const float* b_proj = (const float*)d_in[5];
  const float* view_W = (const float*)d_in[6];
  const float* view_b = (const float*)d_in[7];
  const float* vert_W = (const float*)d_in[8];
  const float* vert_b = (const float*)d_in[9];
  const float* aW1    = (const float*)d_in[10];
  const float* ab1    = (const float*)d_in[11];
  const float* aW2    = (const float*)d_in[12];
  const float* ab2    = (const float*)d_in[13];
  const float* cW1    = (const float*)d_in[14];
  const float* cb1    = (const float*)d_in[15];
  const float* cW2    = (const float*)d_in[16];
  const float* cb2    = (const float*)d_in[17];

  char* ws = (char*)d_ws;
  bf16*  wt_proj = (bf16*)(ws + WS_WT_PROJ);
  bf16*  wt_v0   = (bf16*)(ws + WS_WT_V0);
  bf16*  wt_v1   = (bf16*)(ws + WS_WT_V1);
  bf16*  wt_t0   = (bf16*)(ws + WS_WT_T0);
  bf16*  wt_t1   = (bf16*)(ws + WS_WT_T1);
  float* alphaWc = (float*)(ws + WS_ALPHA_WC);
  float* alphaB  = (float*)(ws + WS_ALPHA_B);
  float* colorWc = (float*)(ws + WS_COLOR_WC);
  float* colorB  = (float*)(ws + WS_COLOR_B);

  // prep: transpose + bf16-convert GEMM weights (stay L2-hot for the fused kernel)
  k_transpose_bf16<<<(256 * 64 + 255) / 256, 256, 0, stream>>>(w_proj, wt_proj, 36, 256, 64);
  k_transpose_bf16<<<(256 * 256 + 255) / 256, 256, 0, stream>>>(view_W, wt_v0, 256, 256, 256);
  k_transpose_bf16<<<(256 * 256 + 255) / 256, 256, 0, stream>>>(view_W + 65536, wt_v1, 256, 256, 256);
  k_transpose_bf16<<<(256 * 256 + 255) / 256, 256, 0, stream>>>(vert_W, wt_t0, 256, 256, 256);
  k_transpose_bf16<<<(256 * 256 + 255) / 256, 256, 0, stream>>>(vert_W + 65536, wt_t1, 256, 256, 256);
  // prep: collapse identity-activation heads
  k_collapse_heads<<<8, 256, 0, stream>>>(aW1, ab1, aW2, ab2, cW1, cb1, cW2, cb2,
                                          alphaWc, alphaB, colorWc, colorB);
  // fused gather + TDM-staged WMMA MLP
  mesh_fused<<<NROWS / ROWS_PER_BLOCK, THREADS, 0, stream>>>(
      verts, barys, views, emb, wt_proj, wt_v0, wt_v1, wt_t0, wt_t1,
      b_proj, view_b, vert_b, alphaWc, alphaB, colorWc, colorB, (float*)d_out);
}
